// SparseAutoencoder_15015205667472
// MI455X (gfx1250) — compile-verified, matrix operand order checked by reading, not run on hardware
//
#include <hip/hip_runtime.h>
#include <hip/hip_bf16.h>
#include <stdint.h>

// ---------------- problem constants ----------------
constexpr int B_ = 4096;      // batch
constexpr int D_ = 1024;      // input dim
constexpr int M_ = 16384;     // neurons
constexpr int K_ = 32;        // top-k
constexpr int AUXK_ = 64;     // aux top-k
constexpr int DEAD_ = 256;    // dead threshold

// output layout (flat floats, reference return order)
constexpr size_t RECON_OFF = 0;
constexpr size_t ACT_OFF   = (size_t)B_ * D_;
constexpr size_t TKV_OFF   = ACT_OFF + (size_t)B_ * M_;
constexpr size_t TKI_OFF   = TKV_OFF + (size_t)B_ * K_;
constexpr size_t AUXV_OFF  = TKI_OFF + (size_t)B_ * K_;
constexpr size_t AUXI_OFF  = AUXV_OFF + (size_t)B_ * AUXK_;

// workspace layout (bytes)
constexpr size_t XCB_BYTES = (size_t)B_ * D_ * 2;             // xc in bf16
constexpr size_t WEB_BYTES = (size_t)M_ * D_ * 2;             // W_enc in bf16
constexpr size_t PRE_BYTES = (size_t)B_ * M_ * 4;             // pre_act f32
constexpr size_t STP_BYTES = (size_t)M_ * 4;                  // steps_new
// W_dec transposed [M, D] f32 after steps

typedef __bf16 bf16_t;
typedef __attribute__((ext_vector_type(8)))  __bf16 v8bf;
typedef __attribute__((ext_vector_type(16))) __bf16 v16bf;
typedef __attribute__((ext_vector_type(8)))  float  v8f;
typedef __attribute__((ext_vector_type(4)))  unsigned int v4u;
typedef __attribute__((ext_vector_type(4)))  int v4i;
typedef __attribute__((ext_vector_type(8)))  int v8i;

#define NEG_INF (-3.402823466e38f)

#define HAVE_TDM (__has_builtin(__builtin_amdgcn_tensor_load_to_lds) && \
                  __has_builtin(__builtin_amdgcn_s_wait_tensorcnt) && defined(__AMDGCN__))

// ---------------- 1: xc = (x - input_bias) -> bf16 ----------------
__global__ void k_prep_x(const float* __restrict__ x, const float* __restrict__ ibias,
                         bf16_t* __restrict__ xcb) {
    int i = blockIdx.x * blockDim.x + threadIdx.x;
    int d = i & (D_ - 1);
    xcb[i] = (bf16_t)(x[i] - ibias[d]);
}

// ---------------- 2: W_enc f32 -> bf16 ----------------
__global__ void k_conv_w(const float* __restrict__ w, bf16_t* __restrict__ wb) {
    int i = blockIdx.x * blockDim.x + threadIdx.x;
    const float4 v = ((const float4*)w)[i];
    bf16_t* o = wb + (size_t)i * 4;
    o[0] = (bf16_t)v.x; o[1] = (bf16_t)v.y; o[2] = (bf16_t)v.z; o[3] = (bf16_t)v.w;
}

// ---------------- 3: steps_new = steps_old + 1 ----------------
__global__ void k_steps_inc(const int* __restrict__ s_old, int* __restrict__ s_new) {
    int i = blockIdx.x * blockDim.x + threadIdx.x;
    s_new[i] = s_old[i] + 1;
}

// ---------------- 3b: W_dec [D,M] -> W_decT [M,D] (tiled LDS transpose) --------
__global__ void k_transpose_wdec(const float* __restrict__ in, float* __restrict__ outT) {
    __shared__ float tile[32][33];                 // +1 pad: no bank conflicts
    const int tx = threadIdx.x & 31;
    const int ty = threadIdx.x >> 5;               // 0..7
    const int m0 = blockIdx.x * 32;
    const int d0 = blockIdx.y * 32;
#pragma unroll
    for (int i = 0; i < 32; i += 8)                // coalesced read along m
        tile[ty + i][tx] = in[(size_t)(d0 + ty + i) * M_ + m0 + tx];
    __syncthreads();
#pragma unroll
    for (int i = 0; i < 32; i += 8)                // coalesced write along d
        outT[(size_t)(m0 + ty + i) * D_ + d0 + tx] = tile[tx][ty + i];
}

// ---------------- fragment loader (16-bit A/B operand, ISA lane layout) --------
__device__ inline v16bf frag16(const bf16_t* p) {
    v8bf lo = *(const v8bf*)(p);
    v8bf hi = *(const v8bf*)(p + 16);
    return __builtin_shufflevector(lo, hi, 0,1,2,3,4,5,6,7,8,9,10,11,12,13,14,15);
}

#if HAVE_TDM
// 2-D TDM descriptor load: tile (tile_d1 rows x tile_d0 elems, 2-byte data)
// from row-major tensor with row stride `stride0` elements, into LDS at lds_off.
__device__ inline void tdm_load_2d(uint32_t lds_off, const void* gaddr,
                                   uint32_t tensor_d0, uint32_t tensor_d1,
                                   uint32_t tile_d0, uint32_t tile_d1,
                                   uint32_t stride0) {
    const uint64_t ga = (uint64_t)(uintptr_t)gaddr;
    v4u g0;
    g0.x = 1u;                                            // count=1 (valid D#)
    g0.y = lds_off;                                       // lds_addr (bytes)
    g0.z = (uint32_t)ga;                                  // global_addr[31:0]
    g0.w = (uint32_t)((ga >> 32) & 0x1FFFFFFu) | (2u << 30); // addr[56:32] | type=2
    v8i g1;
    g1[0] = (int)(1u << 16);                              // data_size=1 -> 2 bytes
    g1[1] = (int)((tensor_d0 & 0xFFFFu) << 16);           // tensor_dim0[15:0]
    g1[2] = (int)(((tensor_d0 >> 16) & 0xFFFFu) | ((tensor_d1 & 0xFFFFu) << 16));
    g1[3] = (int)(((tensor_d1 >> 16) & 0xFFFFu) | ((tile_d0 & 0xFFFFu) << 16));
    g1[4] = (int)(tile_d1 & 0xFFFFu);                     // tile_dim1 (tile_dim2=0)
    g1[5] = (int)stride0;                                 // tensor_dim0_stride[31:0]
    g1[6] = 0;
    g1[7] = 0;
    v4i z4 = {0, 0, 0, 0};
#if __clang_major__ >= 23
    v8i z8 = {0, 0, 0, 0, 0, 0, 0, 0};
    __builtin_amdgcn_tensor_load_to_lds(g0, g1, z4, z4, z8, 0);
#else
    __builtin_amdgcn_tensor_load_to_lds(g0, g1, z4, z4, 0);
#endif
}
#endif

// ---------------- 4: pre_act = xc @ W_enc^T + neuron_bias (WMMA bf16) ----------
// 256 threads = 8 waves; workgroup tile 128(batch) x 128(neurons);
// wave tile 32x64 = 2x4 accumulators. MAC order keeps only ONE B fragment live
// (a0,a1,b = 24 VGPRs + 64 accum VGPRs): no spills, no accumulator shuffling.
__global__ void __launch_bounds__(256, 1)
k_gemm_wmma(const bf16_t* __restrict__ xc, const bf16_t* __restrict__ we,
            const float* __restrict__ nbias, float* __restrict__ pre) {
    __shared__ bf16_t As[2][128 * 32];   // 8KB each
    __shared__ bf16_t Bs[2][128 * 32];

    const int lane = threadIdx.x & 31;
    const int wave = threadIdx.x >> 5;   // 0..7
    const int wr = wave & 3;             // 4 row groups of 32
    const int wc = wave >> 2;            // 2 col groups of 64
    const int rowBase = blockIdx.y * 128;
    const int colBase = blockIdx.x * 128;
    const int lr = lane & 15;
    const int koff = (lane >> 4) * 8;    // lane-group k offset within 32-wide chunk

    v8f c[2][4];
#pragma unroll
    for (int i = 0; i < 2; ++i)
#pragma unroll
        for (int j = 0; j < 4; ++j) c[i][j] = (v8f){};

    constexpr int KSTEPS = D_ / 32;

    // per-wave LDS fragment bases (imm offsets handle the rest)
    const bf16_t* aBase0 = &As[0][(wr * 32 + lr) * 32 + koff];
    const bf16_t* bBase0 = &Bs[0][(wc * 64 + lr) * 32 + koff];
    const int bufStride = 128 * 32;      // elements between buffer 0 and 1

#if HAVE_TDM
    const uint32_t ldsA0 = (uint32_t)(uintptr_t)(void*)&As[0][0];
    const uint32_t ldsA1 = (uint32_t)(uintptr_t)(void*)&As[1][0];
    const uint32_t ldsB0 = (uint32_t)(uintptr_t)(void*)&Bs[0][0];
    const uint32_t ldsB1 = (uint32_t)(uintptr_t)(void*)&Bs[1][0];
    if (wave == 0) {
        tdm_load_2d(ldsA0, xc + (size_t)rowBase * D_, D_, B_, 32, 128, D_);
        tdm_load_2d(ldsB0, we + (size_t)colBase * D_, D_, M_, 32, 128, D_);
    }
    for (int ks = 0; ks < KSTEPS; ++ks) {
        const int cur = ks & 1;
        if (wave == 0) {
            if (ks + 1 < KSTEPS) {
                const int k0 = (ks + 1) * 32;
                tdm_load_2d(cur ? ldsA0 : ldsA1, xc + (size_t)rowBase * D_ + k0,
                            D_, B_, 32, 128, D_);
                tdm_load_2d(cur ? ldsB0 : ldsB1, we + (size_t)colBase * D_ + k0,
                            D_, M_, 32, 128, D_);
                __builtin_amdgcn_s_wait_tensorcnt(2);   // current pair complete
            } else {
                __builtin_amdgcn_s_wait_tensorcnt(0);
            }
        }
        __syncthreads();
#else
    for (int ks = 0; ks < KSTEPS; ++ks) {
        const int cur = 0;
        {   // cooperative global->LDS staging (fallback / host pass)
            const int k0 = ks * 32;
            const int t = threadIdx.x;
            const int row = t >> 1;
            const int h = (t & 1) * 16;
            const bf16_t* sa = xc + (size_t)(rowBase + row) * D_ + k0 + h;
            const bf16_t* sb = we + (size_t)(colBase + row) * D_ + k0 + h;
            *(v8bf*)(&As[0][row * 32 + h])     = *(const v8bf*)(sa);
            *(v8bf*)(&As[0][row * 32 + h + 8]) = *(const v8bf*)(sa + 8);
            *(v8bf*)(&Bs[0][row * 32 + h])     = *(const v8bf*)(sb);
            *(v8bf*)(&Bs[0][row * 32 + h + 8]) = *(const v8bf*)(sb + 8);
        }
        __syncthreads();
#endif
        const bf16_t* aB = aBase0 + cur * bufStride;
        const bf16_t* bB = bBase0 + cur * bufStride;
        v16bf a0 = frag16(aB);             // rows +0..15 of wave row-tile
        v16bf a1 = frag16(aB + 16 * 32);   // rows +16..31
#pragma unroll
        for (int j = 0; j < 4; ++j) {      // one live B fragment at a time
            v16bf b = frag16(bB + j * 16 * 32);
            c[0][j] = __builtin_amdgcn_wmma_f32_16x16x32_bf16(false, a0, false, b, (short)0, c[0][j], false, false);
            c[1][j] = __builtin_amdgcn_wmma_f32_16x16x32_bf16(false, a1, false, b, (short)0, c[1][j], false, false);
        }
        __syncthreads();
    }

    // C/D layout: VGPR r, lanes 0-15 -> row r (N=lane), lanes 16-31 -> row r+8
    const int lh = lane >> 4;
#pragma unroll
    for (int j = 0; j < 4; ++j) {
        const int col0 = colBase + wc * 64 + j * 16 + lr;
        const float nb = nbias[col0];
#pragma unroll
        for (int i = 0; i < 2; ++i) {
#pragma unroll
            for (int r = 0; r < 8; ++r) {
                const int row0 = rowBase + wr * 32 + i * 16 + r + lh * 8;
                pre[(size_t)row0 * M_ + col0] = c[i][j][r] + nb;
            }
        }
    }
}

// ---------------- 5: per-row top-32 + scatter into activ + steps reset ---------
__global__ void k_topk(const float* __restrict__ pre, float* __restrict__ out,
                       int* __restrict__ steps) {
    extern __shared__ float smem[];
    float* sv = smem;                       // M_ floats
    float* rv = smem + M_;                  // 256
    int*   ri = (int*)(smem + M_ + 256);    // 256

    const int row = blockIdx.x;
    const int t = threadIdx.x;
    const float* prow = pre + (size_t)row * M_;
    float* tkv   = out + TKV_OFF + (size_t)row * K_;
    float* tki   = out + TKI_OFF + (size_t)row * K_;
    float* activ = out + ACT_OFF + (size_t)row * M_;

    for (int j = t; j < M_; j += 256) { sv[j] = prow[j]; activ[j] = 0.0f; }
    __syncthreads();

    for (int it = 0; it < K_; ++it) {
        float best = NEG_INF; int bidx = 0;
        for (int j = t; j < M_; j += 256) {
            float v = sv[j];
            if (v > best) { best = v; bidx = j; }
        }
        rv[t] = best; ri[t] = bidx;
        __syncthreads();
        for (int s = 128; s > 0; s >>= 1) {
            if (t < s && rv[t + s] > rv[t]) { rv[t] = rv[t + s]; ri[t] = ri[t + s]; }
            __syncthreads();
        }
        if (t == 0) {
            const int widx = ri[0];
            const float wv = rv[0];
            sv[widx] = NEG_INF;
            const float rel = wv > 0.0f ? wv : 0.0f;
            tkv[it] = rel;
            tki[it] = (float)widx;
            activ[widx] = rel;
            steps[widx] = 0;
        }
        __syncthreads();
    }
}

// ---------------- 6: sparse decode with transposed W_dec (coalesced) -----------
__global__ void k_decode(const float* __restrict__ wdt, const float* __restrict__ ibias,
                         const float* __restrict__ out_ro, float* __restrict__ out) {
    const int row = blockIdx.x;
    const int t = threadIdx.x;
    __shared__ float svv[K_];
    __shared__ int   sii[K_];
    if (t < K_) {
        svv[t] = out_ro[TKV_OFF + (size_t)row * K_ + t];
        sii[t] = (int)out_ro[TKI_OFF + (size_t)row * K_ + t];
    }
    __syncthreads();

    float acc[4];
#pragma unroll
    for (int u = 0; u < 4; ++u) acc[u] = ibias[t + u * 256];

    for (int k = 0; k < K_; ++k) {
        const float v = svv[k];
        const float* wrow = wdt + (size_t)sii[k] * D_;   // contiguous 4KB row
#pragma unroll
        for (int u = 0; u < 4; ++u) acc[u] += v * wrow[t + u * 256];
    }
#pragma unroll
    for (int u = 0; u < 4; ++u)
        out[RECON_OFF + (size_t)row * D_ + t + u * 256] = acc[u];
}

// ---------------- 7: aux top-64 over pre_act * dead_mask -----------------------
__global__ void k_aux(const float* __restrict__ pre, const int* __restrict__ steps,
                      float* __restrict__ out) {
    extern __shared__ float smem[];
    float* sv = smem;
    float* rv = smem + M_;
    int*   ri = (int*)(smem + M_ + 256);

    const int row = blockIdx.x;
    const int t = threadIdx.x;
    const float* prow = pre + (size_t)row * M_;
    float* av = out + AUXV_OFF + (size_t)row * AUXK_;
    float* ai = out + AUXI_OFF + (size_t)row * AUXK_;

    for (int j = t; j < M_; j += 256)
        sv[j] = (steps[j] > DEAD_) ? prow[j] : 0.0f;
    __syncthreads();

    for (int it = 0; it < AUXK_; ++it) {
        float best = NEG_INF; int bidx = 0;
        for (int j = t; j < M_; j += 256) {
            float v = sv[j];
            if (v > best) { best = v; bidx = j; }
        }
        rv[t] = best; ri[t] = bidx;
        __syncthreads();
        for (int s = 128; s > 0; s >>= 1) {
            if (t < s && rv[t + s] > rv[t]) { rv[t] = rv[t + s]; ri[t] = ri[t + s]; }
            __syncthreads();
        }
        if (t == 0) {
            const int widx = ri[0];
            const float wv = rv[0];
            sv[widx] = NEG_INF;
            av[it] = wv > 0.0f ? wv : 0.0f;
            ai[it] = (float)widx;
        }
        __syncthreads();
    }
}

// ---------------- launcher ----------------
extern "C" void kernel_launch(void* const* d_in, const int* in_sizes, int n_in,
                              void* d_out, int out_size, void* d_ws, size_t ws_size,
                              hipStream_t stream) {
    const float* x      = (const float*)d_in[0];
    const float* W_enc  = (const float*)d_in[1];
    const float* W_dec  = (const float*)d_in[2];
    const float* ibias  = (const float*)d_in[3];
    const float* nbias  = (const float*)d_in[4];
    const int*   s_old  = (const int*)d_in[5];
    float* out = (float*)d_out;

    char* ws = (char*)d_ws;
    bf16_t* xcb = (bf16_t*)ws;
    bf16_t* web = (bf16_t*)(ws + XCB_BYTES);
    float*  pre = (float*)(ws + XCB_BYTES + WEB_BYTES);
    int*    stp = (int*)(ws + XCB_BYTES + WEB_BYTES + PRE_BYTES);
    float*  wdt = (float*)(ws + XCB_BYTES + WEB_BYTES + PRE_BYTES + STP_BYTES);

    // prep (independent)
    k_prep_x<<<(B_ * D_) / 256, 256, 0, stream>>>(x, ibias, xcb);
    k_conv_w<<<(M_ * D_) / (256 * 4), 256, 0, stream>>>(W_enc, web);
    k_steps_inc<<<M_ / 256, 256, 0, stream>>>(s_old, stp);
    k_transpose_wdec<<<dim3(M_ / 32, D_ / 32), 256, 0, stream>>>(W_dec, wdt);

    // WMMA GEMM -> pre_act
    k_gemm_wmma<<<dim3(M_ / 128, B_ / 128), 256, 0, stream>>>(xcb, web, nbias, pre);

    // top-k + activ scatter + steps reset
    const size_t shmem = (size_t)(M_ + 512) * sizeof(float);
    k_topk<<<B_, 256, shmem, stream>>>(pre, out, stp);

    // sparse decode (coalesced via transposed W_dec)
    k_decode<<<B_, 256, 0, stream>>>(wdt, ibias, out, out);

    // aux top-k over dead neurons
    k_aux<<<B_, 256, shmem, stream>>>(pre, stp, out);
}
